// CrossAttn_66082366816505
// MI455X (gfx1250) — matrix operand order repaired and checked by reading, not compiled, not run on hardware
//
#include <hip/hip_runtime.h>
#include <cstdint>

// MI455X (gfx1250) flash-attention-style dual cross-attention.
// Compute-bound (~2150 FLOP/byte): bf16 WMMA 16x16x32, fp32 accumulate.
// softmax(-QK^T) folded by storing -Q in LDS. V == Q-source matrix.
// K is staged TRANSPOSED in LDS (Kt[c][n]); A fragments are 2x 16B loads, B fragments 16 strided loads.

typedef __attribute__((ext_vector_type(16))) __bf16 v16bf;
typedef __attribute__((ext_vector_type(8)))  float  v8f;

static constexpr int kB   = 4;
static constexpr int kN   = 4096;   // 64*64
static constexpr int kC   = 256;
static constexpr int kMT  = 64;     // query rows per block
static constexpr int kNT  = 64;     // key rows per inner iteration
static constexpr int kLDA = 264;    // row stride for Qs/Vs (256+8 pad; 528B = 33*16)
static constexpr int kKTL = 72;     // row stride for Kt/Ps  (64+8 pad; 144B = 9*16)

union Frag {                // natural alignment 32 from v16bf member
  v16bf    v;
  uint32_t u[8];
  uint4    q[2];
};

// A-fragment 16x32 bf16 (ISA 7.12.2): lane%16 -> row M; lanes>=16 take +8 K
// offset; VGPRs 0-3 hold k {0..7}, VGPRs 4-7 hold k {16..23} (contiguous!).
// => exactly two 16B LDS loads.
static __device__ inline v16bf load_a_frag(const __bf16* base, int lda, int k0) {
  const int lane  = threadIdx.x & 31;
  const int m     = lane & 15;
  const int khalf = (lane >> 4) << 3;
  const __bf16* p = base + m * lda + k0 + khalf;
  Frag f;
  f.q[0] = *(const uint4*)(p);        // k0+khalf + 0..7
  f.q[1] = *(const uint4*)(p + 16);   // k0+khalf + 16..23
  return f.v;
}

// B-fragment 32x16 bf16 with B[k][n] = src[k0+k][n0+n] (row-major tile).
// lane%16 -> column n; lanes>=16 take +8 K offset; elements 0-7 hold k {0..7},
// elements 8-15 hold k {16..23} (same k order as the A-fragment).
static __device__ inline v16bf load_b_frag(const __bf16* base, int lda, int k0, int n0) {
  const int lane  = threadIdx.x & 31;
  const int khalf = (lane >> 4) << 3;
  const __bf16* p = base + (k0 + khalf) * lda + n0 + (lane & 15);
  Frag f;
#pragma unroll
  for (int s = 0; s < 8; ++s) {
    f.v[s]     = p[s * lda];          // k0+khalf + s
    f.v[s + 8] = p[(s + 16) * lda];   // k0+khalf + 16 + s
  }
  return f.v;
}

__global__ __launch_bounds__(128)
void CrossAttn_66082366816505_kernel(const float* __restrict__ xa,
                                     const float* __restrict__ xb,
                                     const float* __restrict__ gptr,
                                     float* __restrict__ out) {
  extern __shared__ char smem_raw[];
  __bf16* Qs = (__bf16*)smem_raw;        // [kMT][kLDA]   (stores -Q, row-major)
  __bf16* Kt = Qs + kMT * kLDA;          // [kC][kKTL]    (K TRANSPOSED: Kt[c][n])
  __bf16* Vs = Kt + kC * kKTL;           // [kNT][kLDA]   (row-major)
  __bf16* Ps = Vs + kNT * kLDA;          // 4 waves x [16][kKTL]

  const int bx     = blockIdx.x;
  const int tile   = bx & 63;
  const int batch  = (bx >> 6) & 3;
  const int branch = bx >> 8;
  const float* __restrict__ xq = branch ? xb : xa;
  const float* __restrict__ xk = branch ? xa : xb;
  float* __restrict__ obase = out + (size_t)branch * kB * kN * kC;

  const int tid  = threadIdx.x;
  const int wave = tid >> 5;
  const int lane = tid & 31;
  const int hf   = lane >> 4;   // half-wave selector
  const int ln   = lane & 15;

  const float  gamma = *gptr;
  const size_t bofs  = (size_t)batch * kN * kC;
  const int    q0    = tile * kMT;

  // ---- stage -Q tile into LDS (fp32 -> bf16, packed 8B stores) ----
  for (int i = tid; i < kMT * (kC / 4); i += 128) {
    const int row = i >> 6;
    const int c   = (i & 63) << 2;
    const float4 f = *(const float4*)(xq + bofs + (size_t)(q0 + row) * kC + c);
    union { __bf16 h[4]; uint2 u; } pk;
    pk.h[0] = (__bf16)(-f.x); pk.h[1] = (__bf16)(-f.y);
    pk.h[2] = (__bf16)(-f.z); pk.h[3] = (__bf16)(-f.w);
    *(uint2*)(Qs + row * kLDA + c) = pk.u;
  }
  __syncthreads();

  // ---- accumulators: O (16x256 per wave), row stats ----
  v8f o[16];
#pragma unroll
  for (int j = 0; j < 16; ++j)
#pragma unroll
    for (int r = 0; r < 8; ++r) o[j][r] = 0.f;

  float mrun[8], lrun[8];
#pragma unroll
  for (int r = 0; r < 8; ++r) { mrun[r] = -3.0e38f; lrun[r] = 0.f; }

  const __bf16* Qw = Qs + wave * 16 * kLDA;
  __bf16*       Pw = Ps + wave * 16 * kKTL;

  for (int m0 = 0; m0 < kN; m0 += kNT) {
    // ---- stage K tile transposed (Kt[c][n]) and V tile row-major ----
    for (int i = tid; i < kNT * (kC / 4); i += 128) {
      const int row = i >> 6;            // key index n within tile
      const int c   = (i & 63) << 2;
      const size_t g = bofs + (size_t)(m0 + row) * kC + c;
      const float4 fk = *(const float4*)(xk + g);
      const float4 fv = *(const float4*)(xq + g);
      Kt[(c + 0) * kKTL + row] = (__bf16)fk.x;
      Kt[(c + 1) * kKTL + row] = (__bf16)fk.y;
      Kt[(c + 2) * kKTL + row] = (__bf16)fk.z;
      Kt[(c + 3) * kKTL + row] = (__bf16)fk.w;
      union { __bf16 h[4]; uint2 u; } pv;
      pv.h[0] = (__bf16)fv.x; pv.h[1] = (__bf16)fv.y;
      pv.h[2] = (__bf16)fv.z; pv.h[3] = (__bf16)fv.w;
      *(uint2*)(Vs + row * kLDA + c) = pv.u;
      if (m0 + kNT < kN) {                       // global_prefetch_b8 next tile
        __builtin_prefetch(xk + g + (size_t)kNT * kC, 0, 1);
        __builtin_prefetch(xq + g + (size_t)kNT * kC, 0, 1);
      }
    }
    __syncthreads();

    // ---- S = (-Q) K^T : 16x64 per wave, 32 WMMAs ----
    v8f acc[4];
#pragma unroll
    for (int t = 0; t < 4; ++t)
#pragma unroll
      for (int r = 0; r < 8; ++r) acc[t][r] = 0.f;

#pragma unroll
    for (int c = 0; c < 8; ++c) {
      const v16bf aq = load_a_frag(Qw, kLDA, c * 32);
#pragma unroll
      for (int t = 0; t < 4; ++t) {
        const v16bf bk = load_b_frag(Kt, kKTL, c * 32, t * 16);
        acc[t] = __builtin_amdgcn_wmma_f32_16x16x32_bf16(
            false, aq, false, bk, (short)0, acc[t], false, false);
      }
    }

    // ---- online softmax (row stats across lane halves) ----
    float rmax[8];
#pragma unroll
    for (int r = 0; r < 8; ++r) {
      float v = fmaxf(fmaxf(acc[0][r], acc[1][r]), fmaxf(acc[2][r], acc[3][r]));
#pragma unroll
      for (int s = 1; s < 16; s <<= 1) v = fmaxf(v, __shfl_xor(v, s, 32));
      rmax[r] = v;
    }
#pragma unroll
    for (int r = 0; r < 8; ++r) {
      const float mnew = fmaxf(mrun[r], rmax[r]);
      const float corr = __expf(mrun[r] - mnew);
      float ps = 0.f;
#pragma unroll
      for (int t = 0; t < 4; ++t) {
        const float p = __expf(acc[t][r] - mnew);
        ps += p;
        Pw[(r + 8 * hf) * kKTL + t * 16 + ln] = (__bf16)p;  // re-layout via LDS
      }
#pragma unroll
      for (int s = 1; s < 16; s <<= 1) ps += __shfl_xor(ps, s, 32);
      lrun[r] = lrun[r] * corr + ps;
      mrun[r] = mnew;
#pragma unroll
      for (int j = 0; j < 16; ++j) o[j][r] *= corr;
    }

    // ---- O += P @ V : 16x256 per wave, 32 WMMAs ----
#pragma unroll
    for (int kk = 0; kk < 2; ++kk) {
      const v16bf ap = load_a_frag(Pw, kKTL, kk * 32);
#pragma unroll
      for (int j = 0; j < 16; ++j) {
        const v16bf bv = load_b_frag(Vs, kLDA, kk * 32, j * 16);
        o[j] = __builtin_amdgcn_wmma_f32_16x16x32_bf16(
            false, ap, false, bv, (short)0, o[j], false, false);
      }
    }
    __syncthreads();
  }

  // ---- epilogue: out = gamma * (O / l) + x_q ----
  const int rowg = q0 + wave * 16;
#pragma unroll
  for (int j = 0; j < 16; ++j) {
#pragma unroll
    for (int r = 0; r < 8; ++r) {
      const int row = rowg + r + 8 * hf;
      const int col = j * 16 + ln;
      const size_t idx = bofs + (size_t)row * kC + col;
      obase[idx] = fmaf(gamma, o[j][r] / lrun[r], xq[idx]);
    }
  }
}

extern "C" void kernel_launch(void* const* d_in, const int* in_sizes, int n_in,
                              void* d_out, int out_size, void* d_ws, size_t ws_size,
                              hipStream_t stream) {
  (void)in_sizes; (void)n_in; (void)d_ws; (void)ws_size; (void)out_size;
  const float* xa    = (const float*)d_in[0];
  const float* xb    = (const float*)d_in[1];
  const float* gamma = (const float*)d_in[2];
  float* out = (float*)d_out;

  const size_t shmem =
      (size_t)(kMT * kLDA + kC * kKTL + kNT * kLDA + 4 * 16 * kKTL) * sizeof(__bf16);
  dim3 grid(2 * kB * (kN / kMT));   // 512 blocks: 2 branches x 4 batches x 64 q-tiles
  dim3 block(128);                  // 4 wave32s
  CrossAttn_66082366816505_kernel<<<grid, block, shmem, stream>>>(xa, xb, gamma, out);
}
